// PartialEncoder_67903432950537
// MI455X (gfx1250) — matrix operand order repaired and checked by reading, not compiled, not run on hardware
//
#include <hip/hip_runtime.h>
#include <hip/hip_bf16.h>

typedef __attribute__((ext_vector_type(16))) _Float16 v16h;
typedef __attribute__((ext_vector_type(8)))  float    v8f;

// ---------------------------------------------------------------------------
// Generic GEMM: C[M,N] = A[M,K] @ W[K,N] + bias, optional ReLU.
// Requires M % 16 == 0 and bias != nullptr (true for every call here).
// f16 inputs (converted on load), f32 accumulation via v_wmma_f32_16x16x32_f16.
// One wave (32 lanes) per 16x16 output tile. K zero-padded to a multiple of 32
// via clamped (always-valid) addresses + post-load selects, so all global
// loads issue unconditionally (no exec-mask branching in the hot loop).
// ---------------------------------------------------------------------------
__global__ __launch_bounds__(32)
void gemm_bias_wmma(const float* __restrict__ A, const float* __restrict__ W,
                    const float* __restrict__ bias, float* __restrict__ C,
                    int M, int N, int Kd, int relu) {
  const int tm = blockIdx.x, tn = blockIdx.y;
  const int lane = threadIdx.x;
  const int hs = lane >> 4;                 // lane half: 0 or 1
  const int l  = lane & 15;
  const int row  = tm * 16 + l;             // A row (M % 16 == 0 -> valid)
  const int col  = tn * 16 + l;             // logical B/C column
  const int colc = min(col, N - 1);         // clamped (always valid)
  const float* arow = A + (size_t)row * Kd;
  v8f acc = {};
  for (int k0 = 0; k0 < Kd; k0 += 32) {
    const int ka0 = k0 + (hs ? 8 : 0);
    const int ka1 = k0 + (hs ? 24 : 16);
    const int kb0 = k0 + (hs ? 8 : 0);
    const int kb1 = k0 + (hs ? 24 : 16);
    float av0[8], av1[8], bv[16];
#pragma unroll
    for (int i = 0; i < 8; ++i) av0[i] = arow[min(ka0 + i, Kd - 1)];
#pragma unroll
    for (int i = 0; i < 8; ++i) av1[i] = arow[min(ka1 + i, Kd - 1)];
#pragma unroll
    for (int i = 0; i < 16; ++i)
      bv[i] = W[(size_t)min((i < 8 ? kb0 + i : kb1 + i - 8), Kd - 1) * N + colc];
    v16h a, b;
#pragma unroll
    for (int i = 0; i < 8; ++i) {
      a[i]     = (ka0 + i < Kd) ? (_Float16)av0[i] : (_Float16)0.0f;
      a[8 + i] = (ka1 + i < Kd) ? (_Float16)av1[i] : (_Float16)0.0f;
    }
#pragma unroll
    for (int i = 0; i < 16; ++i)
      b[i] = ((i < 8 ? kb0 + i : kb1 + i - 8) < Kd && col < N) ? (_Float16)bv[i] : (_Float16)0.0f;
    acc = __builtin_amdgcn_wmma_f32_16x16x32_f16(false, a, false, b,
                                                 (short)0, acc, false, false);
  }
  if (col < N) {
    const float bvl = bias[col];
    float* cp = C + (size_t)(tm * 16 + (hs ? 8 : 0)) * N + col;
#pragma unroll
    for (int r = 0; r < 8; ++r) {
      float v = acc[r] + bvl;
      if (relu) v = fmaxf(v, 0.0f);
      cp[(size_t)r * N] = v;
    }
  }
}

// ---------------------------------------------------------------------------
// Row-wise LayerNorm (+ optional residual add AFTER norm, + optional ReLU).
// One wave32 per row (coalesced column access), 4 rows per 128-thread block.
// Cross-lane reduction with __shfl_xor (wave32 on gfx1250).
// ---------------------------------------------------------------------------
__global__ __launch_bounds__(128)
void ln_kernel(const float* __restrict__ X, const float* __restrict__ g,
               const float* __restrict__ bt, const float* __restrict__ res,
               float* __restrict__ out, int rows, int nc, int relu) {
  const int lane = threadIdx.x & 31;
  const int r = blockIdx.x * 4 + (threadIdx.x >> 5);
  if (r >= rows) return;
  const float* x = X + (size_t)r * nc;
  float s = 0.0f, ss = 0.0f;
  for (int c = lane; c < nc; c += 32) { float v = x[c]; s += v; ss += v * v; }
#pragma unroll
  for (int m = 16; m >= 1; m >>= 1) {
    s  += __shfl_xor(s, m, 32);
    ss += __shfl_xor(ss, m, 32);
  }
  const float mean = s / nc;
  const float var  = ss / nc - mean * mean;
  const float rs = rsqrtf(var + 1e-5f);
  const float* rp = res ? res + (size_t)r * nc : nullptr;
  float* o = out + (size_t)r * nc;
  for (int c = lane; c < nc; c += 32) {
    float v = (x[c] - mean) * rs * g[c] + bt[c];
    if (rp) v += rp[c];
    if (relu) v = fmaxf(v, 0.0f);
    o[c] = v;
  }
}

// ---------------------------------------------------------------------------
// Brute-force kNN (k=16): one thread per query point, register top-16
// insertion sort over all support points (ascending squared distance).
// ---------------------------------------------------------------------------
__global__ void knn_kernel(const float* __restrict__ q, const float* __restrict__ sup,
                           int* __restrict__ out, int Bn, int S, int N) {
  int t = blockIdx.x * blockDim.x + threadIdx.x;
  if (t >= Bn * S) return;
  const int b = t / S;
  const float* qp = q + (size_t)t * 3;
  const float qx = qp[0], qy = qp[1], qz = qp[2];
  const float* sp = sup + (size_t)b * N * 3;
  float bd[16]; int bi[16];
#pragma unroll
  for (int j = 0; j < 16; ++j) { bd[j] = 3.0e38f; bi[j] = 0; }
  for (int n = 0; n < N; ++n) {
    __builtin_prefetch(sp + (size_t)(n + 64) * 3, 0, 0);
    const float dx = qx - sp[n * 3 + 0];
    const float dy = qy - sp[n * 3 + 1];
    const float dz = qz - sp[n * 3 + 2];
    const float d2 = dx * dx + dy * dy + dz * dz;
    if (d2 < bd[15]) {
#pragma unroll
      for (int j = 15; j >= 0; --j) {
        const bool shift = (j > 0) && (d2 < bd[j - 1]);
        if (shift)            { bd[j] = bd[j - 1]; bi[j] = bi[j - 1]; }
        else if (d2 < bd[j])  { bd[j] = d2;        bi[j] = n;         }
      }
    }
  }
  int* op = out + (size_t)t * 16;
#pragma unroll
  for (int j = 0; j < 16; ++j) op[j] = bi[j];
}

// ---------------------------------------------------------------------------
// Fused pointconv front-end: per (b,s) query point:
//   rel[k]  = support_xyz[idx[k]] - new_xyz            (k=0..15)
//   w[k][:] = 2-layer weight-net MLP(rel[k])           (3->4->4, LN+ReLU each)
//   out[c*4+m] = sum_k feats[idx[k]][c] * w[k][m]      (C <= 128)
// One 64-thread block per query point; nf tile staged in LDS.
// ---------------------------------------------------------------------------
__global__ __launch_bounds__(64)
void pointconv_gather(const float* __restrict__ sxyz, const float* __restrict__ feats,
                      const float* __restrict__ nxyz, const int* __restrict__ idx,
                      const float* __restrict__ W1, const float* __restrict__ b1,
                      const float* __restrict__ g1, const float* __restrict__ bt1,
                      const float* __restrict__ W2, const float* __restrict__ b2,
                      const float* __restrict__ g2, const float* __restrict__ bt2,
                      float* __restrict__ out, int S, int N, int C) {
  __shared__ float wsh[16][4];
  __shared__ int   ish[16];
  __shared__ float nfsh[16 * 128];
  const int bs = blockIdx.x;
  const int b = bs / S;
  const int tid = threadIdx.x;
  const int* ip = idx + (size_t)bs * 16;
  if (tid < 16) {
    const int id = ip[tid];
    ish[tid] = id;
    const float* pp = sxyz + ((size_t)b * N + id) * 3;
    const float* np = nxyz + (size_t)bs * 3;
    const float r0 = pp[0] - np[0], r1 = pp[1] - np[1], r2 = pp[2] - np[2];
    float t[4], u[4];
#pragma unroll
    for (int j = 0; j < 4; ++j)
      t[j] = r0 * W1[j] + r1 * W1[4 + j] + r2 * W1[8 + j] + b1[j];
    float m = 0.25f * (t[0] + t[1] + t[2] + t[3]);
    float v = 0.0f;
#pragma unroll
    for (int j = 0; j < 4; ++j) { float d = t[j] - m; v += d * d; }
    float rs = rsqrtf(0.25f * v + 1e-5f);
#pragma unroll
    for (int j = 0; j < 4; ++j) t[j] = fmaxf((t[j] - m) * rs * g1[j] + bt1[j], 0.0f);
#pragma unroll
    for (int j = 0; j < 4; ++j)
      u[j] = t[0] * W2[j] + t[1] * W2[4 + j] + t[2] * W2[8 + j] + t[3] * W2[12 + j] + b2[j];
    m = 0.25f * (u[0] + u[1] + u[2] + u[3]);
    v = 0.0f;
#pragma unroll
    for (int j = 0; j < 4; ++j) { float d = u[j] - m; v += d * d; }
    rs = rsqrtf(0.25f * v + 1e-5f);
#pragma unroll
    for (int j = 0; j < 4; ++j)
      wsh[tid][j] = fmaxf((u[j] - m) * rs * g2[j] + bt2[j], 0.0f);
  }
  __syncthreads();
  const float* fb = feats + (size_t)b * N * C;
  for (int t2 = tid; t2 < 16 * C; t2 += 64) {
    const int k = t2 / C, c = t2 % C;
    nfsh[t2] = fb[(size_t)ish[k] * C + c];
  }
  __syncthreads();
  float* op = out + (size_t)bs * C * 4;
  for (int c = tid; c < C; c += 64) {
    float a0 = 0, a1 = 0, a2 = 0, a3 = 0;
#pragma unroll
    for (int k = 0; k < 16; ++k) {
      const float f = nfsh[k * C + c];
      a0 += f * wsh[k][0]; a1 += f * wsh[k][1];
      a2 += f * wsh[k][2]; a3 += f * wsh[k][3];
    }
    op[c * 4 + 0] = a0; op[c * 4 + 1] = a1; op[c * 4 + 2] = a2; op[c * 4 + 3] = a3;
  }
}

// concat [rows,3] xyz with [rows,cf] feats -> [rows, 3+cf]
__global__ void concat_kernel(const float* __restrict__ xyz, const float* __restrict__ f,
                              float* __restrict__ out, int rows, int cf) {
  const int nc = 3 + cf;
  int t = blockIdx.x * blockDim.x + threadIdx.x;
  if (t >= rows * nc) return;
  const int r = t / nc, c = t % nc;
  out[t] = (c < 3) ? xyz[(size_t)r * 3 + c] : f[(size_t)r * cf + (c - 3)];
}

// max over the S points of g[B,S,nc] -> out[B,nc]
__global__ void rowmax_kernel(const float* __restrict__ g, float* __restrict__ out,
                              int Bn, int S, int nc) {
  int t = blockIdx.x * blockDim.x + threadIdx.x;
  if (t >= Bn * nc) return;
  const int b = t / nc, c = t % nc;
  float m = -3.0e38f;
  for (int s = 0; s < S; ++s)
    m = fmaxf(m, g[((size_t)b * S + s) * nc + c]);
  out[t] = m;
}

// ---------------------------------------------------------------------------
extern "C" void kernel_launch(void* const* d_in, const int* in_sizes, int n_in,
                              void* d_out, int out_size, void* d_ws, size_t ws_size,
                              hipStream_t stream) {
  auto F = [&](int i) { return (const float*)d_in[i]; };
  const float* xyz[5];
  for (int i = 0; i < 5; ++i) xyz[i] = F(2 * i);  // xyz1,nrm1,xyz2,nrm2,...

  // --- params pytree, flattened in sorted-dict-key order ---
  int p = 10;
  // "global": 2 layers x (W,b,bt,g)
  const float *gW[2], *gB[2], *gBt[2], *gG[2];
  for (int l = 0; l < 2; ++l) { gW[l]=F(p++); gB[l]=F(p++); gBt[l]=F(p++); gG[l]=F(p++); }
  // "layers": 4 x { interp{lin(W,b), ln(bt,g), wn[2x(W,b,bt,g)]},
  //                 res{down(W,b,bt,g), pc_lin(W,b), pc_ln(bt,g), up(W,b,bt,g), wn[2x..]} }
  struct WN { const float *W1,*b1,*bt1,*g1,*W2,*b2,*bt2,*g2; };
  struct Layer {
    const float *linW,*linB,*lnBt,*lnG; WN iwn;
    const float *dW,*dB,*dBt,*dG;
    const float *pcW,*pcB,*pcBt,*pcG;
    const float *uW,*uB,*uBt,*uG; WN rwn;
  } Ls[4];
  for (int l = 0; l < 4; ++l) {
    Layer& L = Ls[l];
    L.linW=F(p++); L.linB=F(p++);
    L.lnBt=F(p++); L.lnG=F(p++);
    L.iwn.W1=F(p++); L.iwn.b1=F(p++); L.iwn.bt1=F(p++); L.iwn.g1=F(p++);
    L.iwn.W2=F(p++); L.iwn.b2=F(p++); L.iwn.bt2=F(p++); L.iwn.g2=F(p++);
    L.dW=F(p++); L.dB=F(p++); L.dBt=F(p++); L.dG=F(p++);
    L.pcW=F(p++); L.pcB=F(p++);
    L.pcBt=F(p++); L.pcG=F(p++);
    L.uW=F(p++); L.uB=F(p++); L.uBt=F(p++); L.uG=F(p++);
    L.rwn.W1=F(p++); L.rwn.b1=F(p++); L.rwn.bt1=F(p++); L.rwn.g1=F(p++);
    L.rwn.W2=F(p++); L.rwn.b2=F(p++); L.rwn.bt2=F(p++); L.rwn.g2=F(p++);
  }
  // "pw": 3 layers x (W,b,bt,g)
  const float *pwW[3], *pwB[3], *pwBt[3], *pwG[3];
  for (int l = 0; l < 3; ++l) { pwW[l]=F(p++); pwB[l]=F(p++); pwBt[l]=F(p++); pwG[l]=F(p++); }

  // --- workspace (floats, bump-allocated) ---
  float* w0 = (float*)d_ws;
  float* F0  = w0; w0 += 524288;    // feats ping
  float* F1  = w0; w0 += 524288;    // feats pong
  float* H   = w0; w0 += 524288;    // interp output h (residual source)
  float* G   = w0; w0 += 1048576;   // gather/einsum output, global activations
  float* T   = w0; w0 += 1048576;   // pre-LN GEMM output
  float* D0  = w0; w0 += 131072;
  float* D1  = w0; w0 += 131072;
  float* Gin = w0; w0 += 530432;    // [2048, 259] global MLP input
  int* IDX1 = (int*)w0; w0 += 262144;
  int* IDX2 = (int*)w0; w0 += 262144;

  auto gemm = [&](const float* A, const float* Wt, const float* bias, float* Cq,
                  int M, int N, int Kd, int relu) {
    dim3 g((M + 15) / 16, (N + 15) / 16);
    gemm_bias_wmma<<<g, 32, 0, stream>>>(A, Wt, bias, Cq, M, N, Kd, relu);
  };
  auto ln = [&](const float* X, const float* gg, const float* bb, const float* res,
                float* O, int rows, int nc, int relu) {
    ln_kernel<<<(rows + 3) / 4, 128, 0, stream>>>(X, gg, bb, res, O, rows, nc, relu);
  };

  // --- pointwise MLP 3->16->16->16 over xyz1 (B*2048 points) ---
  const int Mpw = 16 * 2048;
  gemm(xyz[0], pwW[0], pwB[0], T, Mpw, 16, 3, 0);
  ln(T, pwG[0], pwBt[0], nullptr, F0, Mpw, 16, 1);
  gemm(F0, pwW[1], pwB[1], T, Mpw, 16, 16, 0);
  ln(T, pwG[1], pwBt[1], nullptr, F1, Mpw, 16, 1);
  gemm(F1, pwW[2], pwB[2], T, Mpw, 16, 16, 0);
  ln(T, pwG[2], pwBt[2], nullptr, F0, Mpw, 16, 1);

  // --- 4 PCFE layers ---
  const int NSs[5] = {2048, 1024, 512, 256, 128};
  const int douts[4] = {32, 64, 128, 256};
  const float* featsIn = F0;
  for (int l = 0; l < 4; ++l) {
    const Layer& L = Ls[l];
    const int Nin = NSs[l], S = NSs[l + 1];
    const int cin = (l == 0) ? 16 : douts[l - 1];
    const int cout = douts[l], mid = cout / 4;
    const int rows = 16 * S;
    const float* qx = xyz[l + 1];
    const float* sx = xyz[l];

    knn_kernel<<<(rows + 63) / 64, 64, 0, stream>>>(qx, sx, IDX1, 16, S, Nin);
    pointconv_gather<<<rows, 64, 0, stream>>>(sx, featsIn, qx, IDX1,
        L.iwn.W1, L.iwn.b1, L.iwn.g1, L.iwn.bt1,
        L.iwn.W2, L.iwn.b2, L.iwn.g2, L.iwn.bt2, G, S, Nin, cin);
    gemm(G, L.linW, L.linB, T, rows, cout, cin * 4, 0);
    ln(T, L.lnG, L.lnBt, nullptr, H, rows, cout, 1);

    knn_kernel<<<(rows + 63) / 64, 64, 0, stream>>>(qx, qx, IDX2, 16, S, S);
    gemm(H, L.dW, L.dB, T, rows, mid, cout, 0);
    ln(T, L.dG, L.dBt, nullptr, D0, rows, mid, 1);
    pointconv_gather<<<rows, 64, 0, stream>>>(qx, D0, qx, IDX2,
        L.rwn.W1, L.rwn.b1, L.rwn.g1, L.rwn.bt1,
        L.rwn.W2, L.rwn.b2, L.rwn.g2, L.rwn.bt2, G, S, S, mid);
    gemm(G, L.pcW, L.pcB, T, rows, mid, mid * 4, 0);
    ln(T, L.pcG, L.pcBt, nullptr, D1, rows, mid, 1);
    gemm(D1, L.uW, L.uB, T, rows, cout, mid, 0);

    float* fout = (l == 3) ? (float*)d_out : ((l % 2 == 0) ? F1 : F0);
    ln(T, L.uG, L.uBt, /*residual=*/H, fout, rows, cout, 1);
    featsIn = fout;
  }

  // --- global MLP: concat([xyz5, feats4]) -> 512 (LN+ReLU) -> 512, then max ---
  const int Mg = 16 * 128;
  concat_kernel<<<(Mg * 259 + 255) / 256, 256, 0, stream>>>(
      xyz[4], (const float*)d_out, Gin, Mg, 256);
  gemm(Gin, gW[0], gB[0], T, Mg, 512, 259, 0);
  ln(T, gG[0], gBt[0], nullptr, G, Mg, 512, 1);
  gemm(G, gW[1], gB[1], T, Mg, 512, 512, 0);
  rowmax_kernel<<<(16 * 512 + 255) / 256, 256, 0, stream>>>(
      T, (float*)d_out + 524288, 16, 128, 512);
}
